// SpiralMambaTransformerBlock_70772471104074
// MI455X (gfx1250) — matrix operand order repaired and checked by reading, not compiled, not run on hardware
//
#include <hip/hip_runtime.h>
#include <hip/hip_bf16.h>
#include <math.h>

// ---------------- dims ----------------
#define Bb 8
#define Ls 576
#define Cc 768
#define NHh 12
#define HDd 64
#define MLPd 3072
#define DSTATEd 16
#define DCONVd 4
#define DTRANKd 48
#define Mtok (Bb*Ls)   // 4608

typedef __attribute__((ext_vector_type(16))) __bf16 v16bf;
typedef __attribute__((ext_vector_type(8)))  float  v8f;
typedef __attribute__((__vector_size__(16))) int    v4i;

union FragU { v16bf v; unsigned short u[16]; };

// ---------------- CDNA5 async global->LDS (guarded, sync fallback) ----------------
#if defined(__has_builtin)
#if __has_builtin(__builtin_amdgcn_global_load_async_to_lds_b128) && \
    __has_builtin(__builtin_amdgcn_s_wait_asynccnt)
#define USE_ASYNC 1
#endif
#endif
#ifndef USE_ASYNC
#define USE_ASYNC 0
#endif

__device__ __forceinline__ void cp16(const unsigned short* g, unsigned short* l) {
#if USE_ASYNC
    __builtin_amdgcn_global_load_async_to_lds_b128((v4i*)g, (v4i*)l, 0, 0);
#else
    *(uint4*)l = *(const uint4*)g;
#endif
}
__device__ __forceinline__ void cp_wait_all() {
#if USE_ASYNC
    __builtin_amdgcn_s_wait_asynccnt(0);
#endif
}
__device__ __forceinline__ void cp_wait_group() {   // tolerate one in-flight stage (4 ops/thread)
#if USE_ASYNC
    __builtin_amdgcn_s_wait_asynccnt(4);
#endif
}

__device__ __forceinline__ unsigned short f2bf(float f) {
    unsigned int x = __float_as_uint(f);
    unsigned int r = x + 0x7FFFu + ((x >> 16) & 1u);
    return (unsigned short)(r >> 16);
}
__device__ __forceinline__ v8f v8zero() {
    v8f z = {0.f,0.f,0.f,0.f,0.f,0.f,0.f,0.f};
    return z;
}
__device__ __forceinline__ int amapA(int i, int half) {
    // 16-bit A-matrix 16x32 lane layout: contiguous k = 16*half + i (matches B fill)
    return i + 16 * half;
}
__device__ __forceinline__ float geluf(float x) {
    return 0.5f * x * (1.f + erff(x * 0.70710678118654752f));
}
__device__ __forceinline__ float softplusf(float x) {
    return (x > 20.f) ? x : log1pf(__expf(x));
}
__device__ __forceinline__ float siluf(float x) {
    return x / (1.f + __expf(-x));
}

// ---------------- spiral index builder (1 thread) ----------------
__global__ void build_spiral_kernel(int* idx_cw, int* idx_ccw, int* rev_cw, int* rev_ccw) {
    if (threadIdx.x != 0 || blockIdx.x != 0) return;
    const int Hh = 24, Ww = 24;
    for (int pass = 0; pass < 2; ++pass) {
        int* out = pass ? idx_ccw : idx_cw;
        bool vis[Hh * Ww];
        for (int i = 0; i < Hh * Ww; ++i) vis[i] = false;
        int cnt = 0;
        int top = 0, bottom = Hh - 1, left = 0, right = Ww - 1;
        while (top <= bottom && left <= right) {
            if (pass == 0) {
                for (int j = left; j <= right; ++j)
                    if (!vis[top*Ww+j]) { out[cnt++] = top*Ww+j; vis[top*Ww+j] = true; }
                top++;
                for (int i = top; i <= bottom; ++i)
                    if (!vis[i*Ww+right]) { out[cnt++] = i*Ww+right; vis[i*Ww+right] = true; }
                right--;
                if (top <= bottom) {
                    for (int j = right; j >= left; --j)
                        if (!vis[bottom*Ww+j]) { out[cnt++] = bottom*Ww+j; vis[bottom*Ww+j] = true; }
                    bottom--;
                }
                if (left <= right) {
                    for (int i = bottom; i >= top; --i)
                        if (!vis[i*Ww+left]) { out[cnt++] = i*Ww+left; vis[i*Ww+left] = true; }
                    left++;
                }
            } else {
                for (int j = right; j >= left; --j)
                    if (!vis[top*Ww+j]) { out[cnt++] = top*Ww+j; vis[top*Ww+j] = true; }
                top++;
                for (int i = top; i <= bottom; ++i)
                    if (!vis[i*Ww+left]) { out[cnt++] = i*Ww+left; vis[i*Ww+left] = true; }
                left++;
                if (top <= bottom) {
                    for (int j = left; j <= right; ++j)
                        if (!vis[bottom*Ww+j]) { out[cnt++] = bottom*Ww+j; vis[bottom*Ww+j] = true; }
                    bottom--;
                }
                if (left <= right) {
                    for (int i = bottom; i >= top; --i)
                        if (!vis[i*Ww+right]) { out[cnt++] = i*Ww+right; vis[i*Ww+right] = true; }
                    right--;
                }
            }
        }
    }
    for (int i = 0; i < Ls; ++i) { rev_cw[idx_cw[i]] = i; rev_ccw[idx_ccw[i]] = i; }
}

// ---------------- weight convert + zero-pad: f32[N][K] -> bf16[Npad][Kpad] ----------------
__global__ __launch_bounds__(256)
void convert_pad_kernel(const float* __restrict__ W, unsigned short* __restrict__ Wb,
                        int N, int K, int Npad, int Kpad) {
    int i = blockIdx.x * 256 + threadIdx.x;
    if (i >= Npad * Kpad) return;
    int n = i / Kpad, k = i - n * Kpad;
    float v = (n < N && k < K) ? W[(size_t)n * K + k] : 0.f;
    Wb[i] = f2bf(v);
}

// ---------------- layernorm: 1 wave per row, bf16 output ----------------
__global__ __launch_bounds__(256)
void layernorm_kernel(const float* __restrict__ x, const float* __restrict__ s,
                      const float* __restrict__ b, unsigned short* __restrict__ out, int M) {
    int wid = threadIdx.x >> 5, lane = threadIdx.x & 31;
    int row = blockIdx.x * 8 + wid;
    if (row >= M) return;
    const float* xr = x + (size_t)row * Cc;
    float sum = 0.f;
    for (int c = lane; c < Cc; c += 32) sum += xr[c];
    #pragma unroll
    for (int o = 16; o; o >>= 1) sum += __shfl_xor(sum, o, 32);
    float mu = sum * (1.f / Cc);
    float vs = 0.f;
    for (int c = lane; c < Cc; c += 32) { float d = xr[c] - mu; vs += d * d; }
    #pragma unroll
    for (int o = 16; o; o >>= 1) vs += __shfl_xor(vs, o, 32);
    float inv = rsqrtf(vs * (1.f / Cc) + 1e-6f);
    unsigned short* orow = out + (size_t)row * Cc;
    for (int c = lane; c < Cc; c += 32)
        orow[c] = f2bf((xr[c] - mu) * inv * s[c] + b[c]);
}

// ---------------- WMMA GEMM, bf16 in, async double-buffered LDS staging ----------------
// A: bf16 [4608, lda] (cols [0,K) used, K multiple of 32, staging reads always in-bounds)
// W: bf16 [Npad, ldw] zero-padded. Out (f32 and/or bf16): [4608, N] with col<N mask.
// ACT: 0 none, 1 gelu, 2 softplus
template<int ACT>
__global__ __launch_bounds__(256)
void gemm_bf16_kernel(const unsigned short* __restrict__ A, int lda,
                      const unsigned short* __restrict__ W, int ldw,
                      const float* __restrict__ bias,
                      const float* __restrict__ resid,
                      float* __restrict__ OutF,
                      unsigned short* __restrict__ OutH,
                      int N, int K) {
    __shared__ unsigned short As[2][128][40];
    __shared__ unsigned short Ws[2][128][40];
    const int tid  = threadIdx.x;
    const int bm   = blockIdx.y * 128, bn = blockIdx.x * 128;
    const int wid  = tid >> 5, lane = tid & 31;
    const int wm   = wid >> 1, wn = wid & 1;       // 4 x 2 wave grid, wave tile 32x64
    const int half = lane >> 4, l16 = lane & 15;

    v8f acc[2][4];
    #pragma unroll
    for (int mi = 0; mi < 2; ++mi)
        #pragma unroll
        for (int ni = 0; ni < 4; ++ni) acc[mi][ni] = v8zero();

    const int nstep = K >> 5;
    auto stage = [&](int s) {
        int buf = s & 1, k0 = s << 5;
        #pragma unroll
        for (int c = 0; c < 2; ++c) {              // A tile: 128 x 32, 2 x 16B per thread
            int id = tid * 2 + c, row = id >> 2, q = id & 3;
            cp16(A + (size_t)(bm + row) * lda + k0 + q * 8, &As[buf][row][q * 8]);
        }
        #pragma unroll
        for (int c = 0; c < 2; ++c) {              // W tile: 128 x 32
            int id = tid * 2 + c, row = id >> 2, q = id & 3;
            cp16(W + (size_t)(bn + row) * ldw + k0 + q * 8, &Ws[buf][row][q * 8]);
        }
    };

    stage(0);
    for (int s = 0; s < nstep; ++s) {
        const int buf = s & 1;
        if (s + 1 < nstep) { stage(s + 1); cp_wait_group(); }
        else               { cp_wait_all(); }
        __syncthreads();

        FragU a[2], bfr[4];
        #pragma unroll
        for (int mi = 0; mi < 2; ++mi) {
            int row = wm * 32 + mi * 16 + l16;
            #pragma unroll
            for (int i = 0; i < 16; ++i) a[mi].u[i] = As[buf][row][amapA(i, half)];
        }
        #pragma unroll
        for (int ni = 0; ni < 4; ++ni) {
            int col = wn * 64 + ni * 16 + l16;
            #pragma unroll
            for (int i = 0; i < 16; ++i) bfr[ni].u[i] = Ws[buf][col][16 * half + i];
        }
        #pragma unroll
        for (int mi = 0; mi < 2; ++mi)
            #pragma unroll
            for (int ni = 0; ni < 4; ++ni)
                acc[mi][ni] = __builtin_amdgcn_wmma_f32_16x16x32_bf16(
                    false, a[mi].v, false, bfr[ni].v, (short)0, acc[mi][ni], false, false);
        __syncthreads();
    }

    // epilogue
    #pragma unroll
    for (int mi = 0; mi < 2; ++mi) {
        #pragma unroll
        for (int ni = 0; ni < 4; ++ni) {
            int col = bn + wn * 64 + ni * 16 + l16;
            if (col >= N) continue;
            #pragma unroll
            for (int r = 0; r < 8; ++r) {
                int row = bm + wm * 32 + mi * 16 + r + 8 * half;
                float v = acc[mi][ni][r];
                if (bias) v += bias[col];
                if (ACT == 1) v = geluf(v);
                if (ACT == 2) v = softplusf(v);
                if (resid) v += resid[(size_t)row * N + col];
                if (OutF) OutF[(size_t)row * N + col] = v;
                if (OutH) OutH[(size_t)row * N + col] = f2bf(v);
            }
        }
    }
}

// ---------------- flash attention (bf16 q/k/v, async K/Q staging, WMMA QK^T & PV) ----------------
__global__ __launch_bounds__(128)
void flash_attn_kernel(const unsigned short* __restrict__ q, const unsigned short* __restrict__ k,
                       const unsigned short* __restrict__ v, unsigned short* __restrict__ ctx) {
    __shared__ unsigned short Qs[64][72];
    __shared__ unsigned short Ks[64][72];
    __shared__ unsigned short Vt[64][72];
    __shared__ unsigned short Ps[64][72];
    __shared__ float Sb[64][65];
    __shared__ float mrow[64], lrow[64], arow[64];

    const int tid  = threadIdx.x;
    const int wid  = tid >> 5, lane = tid & 31;
    const int wm   = wid >> 1, wn = wid & 1;
    const int half = lane >> 4, l16 = lane & 15;
    const int b = blockIdx.z, h = blockIdx.y, qb = blockIdx.x;
    const size_t base = (size_t)b * Ls * Cc + (size_t)h * HDd;

    #pragma unroll
    for (int c = 0; c < 4; ++c) {                  // async Q tile 64x64 bf16
        int id = tid * 4 + c, r = id >> 3, qq = id & 7;
        cp16(q + base + (size_t)(qb * 64 + r) * Cc + qq * 8, &Qs[r][qq * 8]);
    }
    if (tid < 64) { mrow[tid] = -1e30f; lrow[tid] = 0.f; }

    v8f o[2][2];
    o[0][0] = v8zero(); o[0][1] = v8zero(); o[1][0] = v8zero(); o[1][1] = v8zero();

    for (int j = 0; j < Ls / 64; ++j) {
        __syncthreads();
        #pragma unroll
        for (int c = 0; c < 4; ++c) {              // async K tile
            int id = tid * 4 + c, r = id >> 3, qq = id & 7;
            cp16(k + base + (size_t)(j * 64 + r) * Cc + qq * 8, &Ks[r][qq * 8]);
        }
        for (int i = tid; i < 64 * 64; i += 128) { // manual V transpose (bf16 copy)
            int r = i >> 6, d = i & 63;
            Vt[d][r] = v[base + (size_t)(j * 64 + r) * Cc + d];
        }
        cp_wait_all();
        __syncthreads();

        // S = (Q K^T) * 0.125
        v8f s[2][2];
        s[0][0] = v8zero(); s[0][1] = v8zero(); s[1][0] = v8zero(); s[1][1] = v8zero();
        #pragma unroll
        for (int k0 = 0; k0 < 64; k0 += 32) {
            FragU a[2], bfr[2];
            #pragma unroll
            for (int mi = 0; mi < 2; ++mi) {
                int row = wm * 32 + mi * 16 + l16;
                #pragma unroll
                for (int i = 0; i < 16; ++i) a[mi].u[i] = Qs[row][k0 + amapA(i, half)];
            }
            #pragma unroll
            for (int ni = 0; ni < 2; ++ni) {
                int col = wn * 32 + ni * 16 + l16;
                #pragma unroll
                for (int i = 0; i < 16; ++i) bfr[ni].u[i] = Ks[col][k0 + 16 * half + i];
            }
            #pragma unroll
            for (int mi = 0; mi < 2; ++mi)
                #pragma unroll
                for (int ni = 0; ni < 2; ++ni)
                    s[mi][ni] = __builtin_amdgcn_wmma_f32_16x16x32_bf16(
                        false, a[mi].v, false, bfr[ni].v, (short)0, s[mi][ni], false, false);
        }
        #pragma unroll
        for (int mi = 0; mi < 2; ++mi)
            #pragma unroll
            for (int ni = 0; ni < 2; ++ni) {
                int col = wn * 32 + ni * 16 + l16;
                #pragma unroll
                for (int r = 0; r < 8; ++r)
                    Sb[wm * 32 + mi * 16 + r + 8 * half][col] = s[mi][ni][r] * 0.125f;
            }
        __syncthreads();

        if (tid < 64) {                            // online softmax stats per row
            float mo = mrow[tid], mx = mo;
            for (int c = 0; c < 64; ++c) mx = fmaxf(mx, Sb[tid][c]);
            float al = __expf(mo - mx);
            float sum = 0.f;
            for (int c = 0; c < 64; ++c) {
                float p = __expf(Sb[tid][c] - mx);
                sum += p;
                Ps[tid][c] = f2bf(p);
            }
            lrow[tid] = lrow[tid] * al + sum;
            mrow[tid] = mx;
            arow[tid] = al;
        }
        __syncthreads();

        #pragma unroll
        for (int mi = 0; mi < 2; ++mi)
            #pragma unroll
            for (int ni = 0; ni < 2; ++ni)
                #pragma unroll
                for (int r = 0; r < 8; ++r)
                    o[mi][ni][r] *= arow[wm * 32 + mi * 16 + r + 8 * half];

        #pragma unroll
        for (int k0 = 0; k0 < 64; k0 += 32) {      // O += P * V
            FragU a[2], bfr[2];
            #pragma unroll
            for (int mi = 0; mi < 2; ++mi) {
                int row = wm * 32 + mi * 16 + l16;
                #pragma unroll
                for (int i = 0; i < 16; ++i) a[mi].u[i] = Ps[row][k0 + amapA(i, half)];
            }
            #pragma unroll
            for (int ni = 0; ni < 2; ++ni) {
                int col = wn * 32 + ni * 16 + l16;
                #pragma unroll
                for (int i = 0; i < 16; ++i) bfr[ni].u[i] = Vt[col][k0 + 16 * half + i];
            }
            #pragma unroll
            for (int mi = 0; mi < 2; ++mi)
                #pragma unroll
                for (int ni = 0; ni < 2; ++ni)
                    o[mi][ni] = __builtin_amdgcn_wmma_f32_16x16x32_bf16(
                        false, a[mi].v, false, bfr[ni].v, (short)0, o[mi][ni], false, false);
        }
    }
    __syncthreads();

    #pragma unroll
    for (int mi = 0; mi < 2; ++mi)
        #pragma unroll
        for (int ni = 0; ni < 2; ++ni) {
            int col = wn * 32 + ni * 16 + l16;
            #pragma unroll
            for (int r = 0; r < 8; ++r) {
                int row = wm * 32 + mi * 16 + r + 8 * half;
                float inv = 1.f / lrow[row];
                ctx[base + (size_t)(qb * 64 + row) * Cc + col] = f2bf(o[mi][ni][r] * inv);
            }
        }
}

// ---------------- depthwise causal conv (spiral order) + silu, f32 + bf16 out ----------------
__global__ __launch_bounds__(256)
void conv_silu_kernel(const float* __restrict__ xz, const int* __restrict__ idx,
                      const float* __restrict__ cw, const float* __restrict__ cb,
                      float* __restrict__ xsc, unsigned short* __restrict__ xscH) {
    int t = blockIdx.x % Ls, b = blockIdx.x / Ls;
    int d = blockIdx.y * 256 + threadIdx.x;
    float acc = cb[d];
    #pragma unroll
    for (int kk = 0; kk < DCONVd; ++kk) {
        int tau = t + kk - (DCONVd - 1);
        if (tau >= 0) {
            int row = idx[tau];
            acc += xz[((size_t)b * Ls + row) * (2 * Cc) + d] * cw[d * DCONVd + kk];
        }
    }
    float sv = siluf(acc);
    size_t o = ((size_t)b * Ls + t) * Cc + d;
    xsc[o] = sv;
    xscH[o] = f2bf(sv);
}

// ---------------- selective scan + D skip + silu(z) gate -> bf16 ----------------
__global__ __launch_bounds__(256)
void scan_kernel(const float* __restrict__ dtb, const float* __restrict__ xsc,
                 const float* __restrict__ dbc, const float* __restrict__ xz,
                 const int* __restrict__ idx, const float* __restrict__ A_log,
                 const float* __restrict__ Dp, unsigned short* __restrict__ mo) {
    int b = blockIdx.x;
    int d = blockIdx.y * 256 + threadIdx.x;
    float An[DSTATEd];
    #pragma unroll
    for (int n = 0; n < DSTATEd; ++n) An[n] = -__expf(A_log[d * DSTATEd + n]);
    float Dd = Dp[d];
    float h[DSTATEd];
    #pragma unroll
    for (int n = 0; n < DSTATEd; ++n) h[n] = 0.f;

    for (int t = 0; t < Ls; ++t) {
        size_t tb = (size_t)b * Ls + t;
        float dt = dtb[tb * Cc + d];
        float xs = xsc[tb * Cc + d];
        const float* bc = dbc + tb * (DTRANKd + 2 * DSTATEd);
        float y = 0.f;
        float dtxs = dt * xs;
        #pragma unroll
        for (int n = 0; n < DSTATEd; ++n) {
            float dA = __expf(dt * An[n]);
            h[n] = dA * h[n] + dtxs * bc[DTRANKd + n];
            y += h[n] * bc[DTRANKd + DSTATEd + n];
        }
        y += xs * Dd;
        float z = xz[((size_t)b * Ls + idx[t]) * (2 * Cc) + Cc + d];
        y *= siluf(z);
        mo[tb * Cc + d] = f2bf(y);
    }
}

// ---------------- gather + concat (undo spirals), bf16 passthrough ----------------
__global__ __launch_bounds__(256)
void gather_concat_kernel(const unsigned short* __restrict__ mo_cw,
                          const unsigned short* __restrict__ mo_ccw,
                          const int* __restrict__ rev_cw, const int* __restrict__ rev_ccw,
                          unsigned short* __restrict__ fused) {
    int s = blockIdx.x % Ls, b = blockIdx.x / Ls;
    int kk = blockIdx.y * 256 + threadIdx.x;
    unsigned short val;
    if (kk < Cc) val = mo_cw[((size_t)b * Ls + rev_cw[s]) * Cc + kk];
    else         val = mo_ccw[((size_t)b * Ls + rev_ccw[s]) * Cc + (kk - Cc)];
    fused[((size_t)b * Ls + s) * (2 * Cc) + kk] = val;
}

// ---------------- host-side launch ----------------
extern "C" void kernel_launch(void* const* d_in, const int* in_sizes, int n_in,
                              void* d_out, int out_size, void* d_ws, size_t ws_size,
                              hipStream_t stream) {
    (void)in_sizes; (void)n_in; (void)out_size; (void)ws_size;
    const float* x      = (const float*)d_in[0];
    const float* bq     = (const float*)d_in[2];
    const float* bk     = (const float*)d_in[4];
    const float* bv     = (const float*)d_in[6];
    const float* bo     = (const float*)d_in[8];
    const float* mb1    = (const float*)d_in[10];
    const float* mb2    = (const float*)d_in[12];
    const float* fb1    = (const float*)d_in[32];
    const float* fb2    = (const float*)d_in[34];
    const float* attn_s = (const float*)d_in[35];
    const float* attn_b = (const float*)d_in[36];
    const float* ffn_s  = (const float*)d_in[37];
    const float* ffn_b  = (const float*)d_in[38];
    const float* mam_s  = (const float*)d_in[39];
    const float* mam_b  = (const float*)d_in[40];
    float* out = (float*)d_out;

    size_t off = 0;
    auto alloc = [&](size_t bytes) -> void* {
        off = (off + 255) & ~(size_t)255;
        void* p = (char*)d_ws + off;
        off += bytes;
        return p;
    };
    typedef unsigned short u16;
    const size_t MC = (size_t)Mtok * Cc;

    int* idxb    = (int*)alloc(4 * Ls * sizeof(int));
    u16* xnH     = (u16*)alloc(MC * 2);
    u16* qH      = (u16*)alloc(MC * 2);
    u16* kH      = (u16*)alloc(MC * 2);
    u16* vH      = (u16*)alloc(MC * 2);
    u16* ctxH    = (u16*)alloc(MC * 2);
    float* x1    = (float*)alloc(MC * 4);
    float* xz    = (float*)alloc(MC * 2 * 4);
    float* xsc   = (float*)alloc(MC * 4);
    u16* xscH    = (u16*)alloc(MC * 2);
    float* dbc   = (float*)alloc((size_t)Mtok * 80 * 4);
    u16* dbcH    = (u16*)alloc((size_t)Mtok * 80 * 2);
    float* dtb   = (float*)alloc(MC * 4);
    u16* mobH    = (u16*)alloc(MC * 2);
    u16* moPH_cw = (u16*)alloc(MC * 2);
    u16* moPH_cc = (u16*)alloc(MC * 2);
    u16* fusedH  = (u16*)alloc(MC * 2 * 2);
    u16* fhH     = (u16*)alloc(MC * 2);
    float* x2    = (float*)alloc(MC * 4);
    u16* h1H     = (u16*)alloc((size_t)Mtok * MLPd * 2);

    int* IDX_CW  = idxb;
    int* IDX_CCW = idxb + Ls;
    int* REV_CW  = idxb + 2 * Ls;
    int* REV_CCW = idxb + 3 * Ls;

    // bf16 zero-padded weights
    auto wpad = [&](const float* W, int N, int K, int Npad, int Kpad) -> const u16* {
        u16* dst = (u16*)alloc((size_t)Npad * Kpad * 2);
        int total = Npad * Kpad;
        convert_pad_kernel<<<(total + 255) / 256, 256, 0, stream>>>(W, dst, N, K, Npad, Kpad);
        return dst;
    };
    const u16* WqH = wpad((const float*)d_in[1], Cc, Cc, Cc, Cc);
    const u16* WkH = wpad((const float*)d_in[3], Cc, Cc, Cc, Cc);
    const u16* WvH = wpad((const float*)d_in[5], Cc, Cc, Cc, Cc);
    const u16* WoH = wpad((const float*)d_in[7], Cc, Cc, Cc, Cc);
    const u16* mW1H = wpad((const float*)d_in[9],  MLPd, Cc, MLPd, Cc);
    const u16* mW2H = wpad((const float*)d_in[11], Cc, MLPd, Cc, MLPd);
    const u16* fW1H = wpad((const float*)d_in[31], Cc, 2 * Cc, Cc, 2 * Cc);
    const u16* fW2H = wpad((const float*)d_in[33], Cc, Cc, Cc, Cc);
    const u16* br_inH[2];  const u16* br_xpH[2];  const u16* br_dtH[2];  const u16* br_outH[2];
    for (int br = 0; br < 2; ++br) {
        int pb = 13 + br * 9;
        br_inH[br]  = wpad((const float*)d_in[pb + 0], 2 * Cc, Cc, 2 * Cc, Cc);
        br_xpH[br]  = wpad((const float*)d_in[pb + 3], 80, Cc, 128, Cc);
        br_dtH[br]  = wpad((const float*)d_in[pb + 4], Cc, DTRANKd, Cc, 64);
        br_outH[br] = wpad((const float*)d_in[pb + 8], Cc, Cc, Cc, Cc);
    }

    // gemm wrapper: K must be padded to 32, Npad padded to 128
    auto gemm = [&](int act, const u16* A, int lda, const u16* W, int ldw,
                    const float* bias, const float* resid, float* OF, u16* OH,
                    int N, int Kp, int Npad) {
        dim3 g(Npad / 128, Mtok / 128), blk(256);
        if (act == 0)      gemm_bf16_kernel<0><<<g, blk, 0, stream>>>(A, lda, W, ldw, bias, resid, OF, OH, N, Kp);
        else if (act == 1) gemm_bf16_kernel<1><<<g, blk, 0, stream>>>(A, lda, W, ldw, bias, resid, OF, OH, N, Kp);
        else               gemm_bf16_kernel<2><<<g, blk, 0, stream>>>(A, lda, W, ldw, bias, resid, OF, OH, N, Kp);
    };

    // 0. spiral tables
    build_spiral_kernel<<<1, 1, 0, stream>>>(IDX_CW, IDX_CCW, REV_CW, REV_CCW);

    // 1. attention
    layernorm_kernel<<<Mtok / 8, 256, 0, stream>>>(x, attn_s, attn_b, xnH, Mtok);
    gemm(0, xnH, Cc, WqH, Cc, bq, nullptr, nullptr, qH, Cc, Cc, Cc);
    gemm(0, xnH, Cc, WkH, Cc, bk, nullptr, nullptr, kH, Cc, Cc, Cc);
    gemm(0, xnH, Cc, WvH, Cc, bv, nullptr, nullptr, vH, Cc, Cc, Cc);
    flash_attn_kernel<<<dim3(Ls / 64, NHh, Bb), 128, 0, stream>>>(qH, kH, vH, ctxH);
    gemm(0, ctxH, Cc, WoH, Cc, bo, x, x1, nullptr, Cc, Cc, Cc);

    // 2. spiral mamba
    layernorm_kernel<<<Mtok / 8, 256, 0, stream>>>(x1, mam_s, mam_b, xnH, Mtok);
    for (int br = 0; br < 2; ++br) {
        int pb = 13 + br * 9;
        const float* conv_w = (const float*)d_in[pb + 1];
        const float* conv_b = (const float*)d_in[pb + 2];
        const float* dt_b   = (const float*)d_in[pb + 5];
        const float* A_log  = (const float*)d_in[pb + 6];
        const float* Dp     = (const float*)d_in[pb + 7];
        const int* IDX = br ? IDX_CCW : IDX_CW;
        u16* moPH = br ? moPH_cc : moPH_cw;

        gemm(0, xnH, Cc, br_inH[br], Cc, nullptr, nullptr, xz, nullptr, 2 * Cc, Cc, 2 * Cc);
        conv_silu_kernel<<<dim3(Bb * Ls, Cc / 256), 256, 0, stream>>>(xz, IDX, conv_w, conv_b, xsc, xscH);
        gemm(0, xscH, Cc, br_xpH[br], Cc, nullptr, nullptr, dbc, dbcH, 80, Cc, 128);
        gemm(2, dbcH, 80, br_dtH[br], 64, dt_b, nullptr, dtb, nullptr, Cc, 64, Cc);
        scan_kernel<<<dim3(Bb, Cc / 256), 256, 0, stream>>>(dtb, xsc, dbc, xz, IDX, A_log, Dp, mobH);
        gemm(0, mobH, Cc, br_outH[br], Cc, nullptr, nullptr, nullptr, moPH, Cc, Cc, Cc);
    }
    gather_concat_kernel<<<dim3(Bb * Ls, 2 * Cc / 256), 256, 0, stream>>>(
        moPH_cw, moPH_cc, REV_CW, REV_CCW, fusedH);
    gemm(1, fusedH, 2 * Cc, fW1H, 2 * Cc, fb1, nullptr, nullptr, fhH, Cc, 2 * Cc, Cc);
    gemm(0, fhH, Cc, fW2H, Cc, fb2, x1, x2, nullptr, Cc, Cc, Cc);

    // 3. MLP
    layernorm_kernel<<<Mtok / 8, 256, 0, stream>>>(x2, ffn_s, ffn_b, xnH, Mtok);
    gemm(1, xnH, Cc, mW1H, Cc, mb1, nullptr, nullptr, h1H, MLPd, Cc, MLPd);
    gemm(0, h1H, MLPd, mW2H, MLPd, mb2, x2, out, nullptr, Cc, MLPd, Cc);
}
